// SA_Block_71494025609721
// MI455X (gfx1250) — matrix operand order repaired and checked by reading, not compiled, not run on hardware
//
#include <hip/hip_runtime.h>

typedef __bf16 bf16_t;
typedef __attribute__((ext_vector_type(16))) __bf16 v16bf;
typedef __attribute__((ext_vector_type(8)))  __bf16 v8bf;
typedef __attribute__((ext_vector_type(8)))  float  v8f;
typedef __attribute__((ext_vector_type(4)))  float  v4f;

constexpr int kB = 8;     // batch
constexpr int kC = 256;   // channels
constexpr int kN = 4096;  // H*W
constexpr int kD = 32;    // C/8 (q,k dim) == WMMA K for bf16

static __device__ __forceinline__ v8f wmma_bf16(v16bf a, v16bf b, v8f c) {
  // D = A(16x32 bf16) * B(32x16 bf16) + C(16x16 f32)
  return __builtin_amdgcn_wmma_f32_16x16x32_bf16(
      false, a, false, b, (short)0, c, false, false);
}

// A-operand (16x32 bf16): lane holds row m = lane%16; kbA = (lane>=16)?8:0.
// elements [0..7] = row[kbA+e], [8..15] = row[kbA+16+(e-8)]
static __device__ __forceinline__ v16bf load_rowA(const bf16_t* row, int kbA) {
  union { v16bf v; v8bf h[2]; } u;
  u.h[0] = *(const v8bf*)(row + kbA);
  u.h[1] = *(const v8bf*)(row + kbA + 16);
  return u.v;
}

// B-operand (32x16 bf16): lane holds column n = lane%16; kbB = (lane>=16)?16:0.
// elements [0..7] = col[kbB/2 + e], [8..15] = col[kbB/2 + 16 + (e-8)]
static __device__ __forceinline__ v16bf load_colB(const bf16_t* col, int kbB) {
  union { v16bf v; v8bf h[2]; } u;
  u.h[0] = *(const v8bf*)(col + kbB / 2);
  u.h[1] = *(const v8bf*)(col + kbB / 2 + 16);
  return u.v;
}

// Async global->LDS copy, 16B per lane, tracked by ASYNCcnt (CDNA5 path).
static __device__ __forceinline__ void async_b128(unsigned lds_off, const void* g) {
  unsigned long long ga = (unsigned long long)(uintptr_t)g;
  asm volatile("global_load_async_to_lds_b128 %0, %1, off"
               :: "v"(lds_off), "v"(ga) : "memory");
}
static __device__ __forceinline__ void wait_async0() {
  asm volatile("s_wait_asynccnt 0" ::: "memory");
}

// max with lane^XORM partner via ds_swizzle immediate (group-of-32 XOR mode)
template <int XORM>
static __device__ __forceinline__ float maxswz(float v) {
  int r = __builtin_amdgcn_ds_swizzle(__float_as_int(v), (XORM << 10) | 0x1f);
  return fmaxf(v, __int_as_float(r));
}

// ---------------------------------------------------------------------------
// Projection GEMM: out[o,i] = sum_c W[o,c]*x[b,c,i] + bias[o], cast to bf16.
// One wave = OG stacked 16(o) x 16(i) tiles sharing one x B-tile per k-step,
// so x L2 traffic and the f32->bf16 conversion work drop by OG.
// TR=1 -> out[b][i][M]  (Q,K: [B][N][32]);  TR=0 -> out[b][o][N]  (V: [B][C][N])
// ---------------------------------------------------------------------------
template <int M, bool TR, int OG>
__global__ __launch_bounds__(32)
void proj_kernel(const float* __restrict__ x, const float* __restrict__ W,
                 const float* __restrict__ bias, bf16_t* __restrict__ out) {
  const int lane = threadIdx.x & 31;
  const int hw   = lane >> 4;
  const int lm   = lane & 15;
  const int kbA  = hw * 8;
  const int kbB  = hw * 16;

  constexpr int groupsO = M / 16 / OG;
  int t  = blockIdx.x;
  const int it = t % (kN / 16); t /= (kN / 16);
  const int og = t % groupsO;   t /= groupsO;
  const int b  = t;
  const int o0 = og * OG * 16, i0 = it * 16;

  const float* xb = x + (size_t)b * kC * kN;
  v8f acc[OG];
#pragma unroll
  for (int g = 0; g < OG; ++g) acc[g] = v8f{};

  for (int kc = 0; kc < kC; kc += 32) {
    // one shared B tile = x[kc+K][i0+n]; consecutive lm -> coalesced 64B rows
    const float* xcol = xb + (size_t)(kc + kbB) * kN + (i0 + lm);
    union { v16bf v; bf16_t e[16]; } bmat;
#pragma unroll
    for (int e = 0; e < 16; ++e) bmat.e[e] = (bf16_t)xcol[(size_t)e * kN];
#pragma unroll
    for (int g = 0; g < OG; ++g) {
      const float* wrow = W + (size_t)(o0 + g * 16 + lm) * kC + kc;
      union { v16bf v; bf16_t e[16]; } a;
#pragma unroll
      for (int e = 0; e < 8; ++e)  a.e[e] = (bf16_t)wrow[kbB + e];
#pragma unroll
      for (int e = 8; e < 16; ++e) a.e[e] = (bf16_t)wrow[kbB + e];
      acc[g] = wmma_bf16(a.v, bmat.v, acc[g]);
    }
  }

#pragma unroll
  for (int g = 0; g < OG; ++g)
#pragma unroll
    for (int r = 0; r < 8; ++r) {
      const int o = o0 + g * 16 + r + hw * 8;
      const int i = i0 + lm;
      const float val = acc[g][r] + bias[o];
      if (TR) out[((size_t)b * kN + i) * M + o] = (bf16_t)val;
      else    out[((size_t)b * M + o) * kN + i] = (bf16_t)val;
    }
}

// ---------------------------------------------------------------------------
// Fused flash-attention. Block = 256 threads = 8 waves = 4 query tiles (64
// rows) x 2 channel halves, one batch image. K (32x32) and V (32x256) chunks
// async-staged into double-buffered LDS by the whole block (copy overlaps
// compute via ASYNCcnt). Softmax uses a tile-uniform running max (exact: the
// subtracted constant cancels per row), the denominator is a P x ones WMMA,
// and rescaling is skipped once the running max saturates (alpha == 1).
// ---------------------------------------------------------------------------
__global__ __launch_bounds__(256)
void attn_kernel(const bf16_t* __restrict__ Qb, const bf16_t* __restrict__ Kb,
                 const bf16_t* __restrict__ Vb, const float* __restrict__ x,
                 const float* __restrict__ gamma, float* __restrict__ out) {
  __shared__ bf16_t vbuf[2][kC * 32];                      // 2 x 16 KB
  __shared__ bf16_t kbuf[2][32 * kD];                      // 2 x  2 KB
  __shared__ alignas(16) unsigned char wscratch[8][1024];  // per-wave P/O tile

  const int wave = threadIdx.x >> 5;
  const int lane = threadIdx.x & 31;
  const int hw   = lane >> 4;
  const int lm   = lane & 15;
  const int kbA  = hw * 8;
  const int kbB  = hw * 16;

  const int b   = blockIdx.x >> 6;        // 8 batches
  const int itg = blockIdx.x & 63;        // group of 4 query tiles
  const int ti  = wave >> 1;              // i-tile within group
  const int ch  = wave & 1;               // channel half
  const int i0  = (itg * 4 + ti) * 16;
  const int c0  = ch * 128;

  const bf16_t* kbase = Kb + (size_t)b * kN * kD;
  const bf16_t* vbase = Vb + (size_t)b * kC * kN;

  // Q A-operand: Qb is [B][N][32], one row per lane, two 16B loads.
  const v16bf qa = load_rowA(Qb + ((size_t)b * kN + (i0 + lm)) * kD, kbA);

  // all-ones B operand: rowsum(P) = P x ones on the WMMA unit
  union { v16bf v; bf16_t e[16]; } ones;
#pragma unroll
  for (int e = 0; e < 16; ++e) ones.e[e] = (bf16_t)1.0f;

  v8f accv[8];
#pragma unroll
  for (int ct = 0; ct < 8; ++ct) accv[ct] = v8f{};
  v8f lsum = {};
  float mprev = -3.0e38f;

  // cooperative async stage of one 32-column chunk into buffer `buf`
  auto stage = [&](int j0, int buf) {
    const unsigned koff = (unsigned)(uintptr_t)(&kbuf[buf][0]);
    const unsigned voff = (unsigned)(uintptr_t)(&vbuf[buf][0]);
    const int seg = (lane & 3) * 16;                  // 16B segment in 64B row
    if (wave < 4) {                                   // K: 32 rows x 64B
      const int row = wave * 8 + (lane >> 2);
      async_b128(koff + row * 64 + seg,
                 (const char*)(kbase + (size_t)(j0 + row) * kD) + seg);
    }
#pragma unroll
    for (int q = 0; q < 4; ++q) {                     // V: 256 rows x 64B
      const int c = (wave * 4 + q) * 8 + (lane >> 2);
      async_b128(voff + c * 64 + seg,
                 (const char*)(vbase + (size_t)c * kN + j0) + seg);
    }
  };

  stage(0, 0);
  int cur = 0;

  for (int j0 = 0; j0 < kN; j0 += 32) {
    wait_async0();        // my async writes into buf `cur` are done
    __syncthreads();      // everyone's are; prior reads of buf `cur^1` retired
    if (j0 + 32 < kN) stage(j0 + 32, cur ^ 1);  // overlaps with compute below

    // S = q.k for 32 key columns (two WMMA tiles, K-dim = 32 = kD)
    const bf16_t* kc = &kbuf[cur][0];
    const v16bf kb0 = load_colB(kc + (size_t)lm * kD, kbB);
    const v16bf kb1 = load_colB(kc + (size_t)(16 + lm) * kD, kbB);
    const v8f z = {};
    v8f s0 = wmma_bf16(qa, kb0, z);
    v8f s1 = wmma_bf16(qa, kb1, z);

    // tile-uniform online max: 15 VALU max + 5 ds_swizzle, one scalar alpha
    float mt = fmaxf(s0[0], s1[0]);
#pragma unroll
    for (int r = 1; r < 8; ++r) mt = fmaxf(mt, fmaxf(s0[r], s1[r]));
    mt = maxswz<1>(mt);
    mt = maxswz<2>(mt);
    mt = maxswz<4>(mt);
    mt = maxswz<8>(mt);
    mt = maxswz<16>(mt);
    const float mnew  = fmaxf(mprev, mt);
    const float alpha = __expf(mprev - mnew);
    mprev = mnew;

    float p0[8], p1[8];
#pragma unroll
    for (int r = 0; r < 8; ++r) {
      p0[r] = __expf(s0[r] - mnew);
      p1[r] = __expf(s1[r] - mnew);
    }
    // rescale only when the running max actually moved (alpha is wave-uniform)
    if (alpha < 1.0f) {
#pragma unroll
      for (int ct = 0; ct < 8; ++ct)
#pragma unroll
        for (int r = 0; r < 8; ++r) accv[ct][r] *= alpha;
#pragma unroll
      for (int r = 0; r < 8; ++r) lsum[r] *= alpha;
    }

    // transpose P: D-layout (row r+8*hw, col lm) -> LDS -> A-layout (per-wave)
    bf16_t* pb = (bf16_t*)wscratch[wave];
#pragma unroll
    for (int r = 0; r < 8; ++r) {
      const int row = r + hw * 8;
      pb[row * 32 + lm]      = (bf16_t)p0[r];
      pb[row * 32 + 16 + lm] = (bf16_t)p1[r];
    }
    __builtin_amdgcn_wave_barrier();   // cross-lane LDS dep within the wave
    const v16bf pa = load_rowA(pb + lm * 32, kbA);
    __builtin_amdgcn_wave_barrier();

    // denominator: lsum += P x ones (row-sum on the matrix unit, D-layout)
    lsum = wmma_bf16(pa, ones.v, lsum);

    // O += P * V  (one WMMA per 16-channel tile, K-dim = 32 j-columns)
#pragma unroll
    for (int ct = 0; ct < 8; ++ct) {
      const v16bf vb = load_colB(&vbuf[cur][(size_t)(c0 + ct * 16 + lm) * 32], kbB);
      accv[ct] = wmma_bf16(pa, vb, accv[ct]);
    }
    cur ^= 1;
  }

  // epilogue: out = gamma*(O/l) + x, transposed through LDS for 32B/lane IO
  const float gm = gamma[0];
  float inv[8];
#pragma unroll
  for (int r = 0; r < 8; ++r) inv[r] = 1.0f / lsum[r];

  float* ob = (float*)wscratch[wave];       // 16(c) x 16(i) f32 tile
  const int n2 = lane >> 1, h2 = lane & 1;  // writer: channel-row, i-half
#pragma unroll
  for (int ct = 0; ct < 8; ++ct) {
#pragma unroll
    for (int r = 0; r < 8; ++r)
      ob[lm * 16 + (r + hw * 8)] = gm * accv[ct][r] * inv[r];
    __builtin_amdgcn_wave_barrier();
    const int c = c0 + ct * 16 + n2;
    const int i = i0 + h2 * 8;
    const size_t idx = ((size_t)b * kC + c) * kN + i;
    const v4f* src = (const v4f*)&ob[n2 * 16 + h2 * 8];
    const v4f* xr  = (const v4f*)&x[idx];
    v4f*       dst = (v4f*)&out[idx];
    dst[0] = src[0] + xr[0];
    dst[1] = src[1] + xr[1];
    __builtin_amdgcn_wave_barrier();
  }
}

extern "C" void kernel_launch(void* const* d_in, const int* in_sizes, int n_in,
                              void* d_out, int out_size, void* d_ws, size_t ws_size,
                              hipStream_t stream) {
  const float* x     = (const float*)d_in[0];
  const float* Wq    = (const float*)d_in[1];
  const float* bq    = (const float*)d_in[2];
  const float* Wk    = (const float*)d_in[3];
  const float* bk    = (const float*)d_in[4];
  const float* Wv    = (const float*)d_in[5];
  const float* bv    = (const float*)d_in[6];
  const float* gamma = (const float*)d_in[7];
  float* out = (float*)d_out;

  // workspace: Q [B][N][32] bf16, K [B][N][32] bf16, V [B][C][N] bf16 (~21 MB)
  bf16_t* Qb = (bf16_t*)d_ws;
  bf16_t* Kb = Qb + (size_t)kB * kN * kD;
  bf16_t* Vb = Kb + (size_t)kB * kN * kD;

  const int tilesN = kN / 16;
  // Q,K: 2 o-tiles per wave; V: 4 o-tiles per wave sharing one x B-tile
  proj_kernel<kD, true,  2><<<kB * 1 * tilesN, 32, 0, stream>>>(x, Wq, bq, Qb);
  proj_kernel<kD, true,  2><<<kB * 1 * tilesN, 32, 0, stream>>>(x, Wk, bk, Kb);
  proj_kernel<kC, false, 4><<<kB * 4 * tilesN, 32, 0, stream>>>(x, Wv, bv, Vb);

  // block = (batch, 4 query tiles); 8 waves = 4 i-tiles x 2 channel halves
  attn_kernel<<<kB * 64, 256, 0, stream>>>(Qb, Kb, Vb, x, gamma, out);
}
